// Attention_4569845203330
// MI455X (gfx1250) — matrix operand order repaired and checked by reading, not compiled, not run on hardware
//
#include <hip/hip_runtime.h>
#include <hip/hip_bf16.h>

typedef __attribute__((ext_vector_type(16))) _Float16     v16h;
typedef __attribute__((ext_vector_type(8)))  _Float16     v8h;
typedef __attribute__((ext_vector_type(8)))  float        v8f;
typedef __attribute__((ext_vector_type(4)))  float        v4f;
typedef __attribute__((ext_vector_type(4)))  unsigned int v4u;
typedef __attribute__((ext_vector_type(8)))  int          v8i;
typedef __attribute__((ext_vector_type(4)))  int          v4i;

#define DIM   768
#define NH    12
#define HD    64
#define SEQ   1024
#define BATCH 16
#define MTOK  (BATCH*SEQ)        // 16384 tokens
#define E3    (3*DIM)            // 2304
#define QK_SCALE 0.125f          // HEAD_DIM^-0.5

// ---------------------------------------------------------------------------
// WMMA helper: D = A(16x32 f16) * B(32x16 f16) + C(16x16 f32)
// ---------------------------------------------------------------------------
static __device__ __forceinline__ v8f wmma16(v16h a, v16h b, v8f c) {
    return __builtin_amdgcn_wmma_f32_16x16x32_f16(
        false, a, false, b, (short)0, c, false, false);
}

// ---------------------------------------------------------------------------
// Tensor Data Mover: 2D f16 tile (tile_rows x tile_cols elements) from a
// row-major tensor (row stride in elements) into LDS (row-major, packed).
// Descriptor layout per CDNA5 ISA 8.3/8.4 (D# group0 128b, group1 256b).
// Issue from ONE wave per block; TDM ignores EXEC and issues once per wave.
// ---------------------------------------------------------------------------
static __device__ __forceinline__ void tdm_load_2d(
        _Float16* lds_dst, const _Float16* gsrc,
        unsigned tile_cols, unsigned tile_rows,
        unsigned tensor_cols, unsigned tensor_rows, unsigned row_stride) {
    unsigned long long ga = (unsigned long long)(uintptr_t)gsrc;
    unsigned lds_addr = (unsigned)(uintptr_t)lds_dst;   // low 32 bits = LDS byte offset

    v4u g0;
    g0[0] = 1u;                                          // count=1, user mode
    g0[1] = lds_addr;                                    // [63:32] lds_addr
    g0[2] = (unsigned)(ga & 0xFFFFFFFFu);                // [95:64] global_addr lo
    g0[3] = (unsigned)((ga >> 32) & 0x01FFFFFFu)         // [120:96] global_addr hi
          | (2u << 30);                                  // [127:126] type=2 (image)

    v8i g1;
    g1[0] = (int)(1u << 16);                             // data_size=1 (2 bytes)
    g1[1] = (int)((tensor_cols & 0xFFFFu) << 16);        // tensor_dim0[15:0] (bits 63:48)
    g1[2] = (int)(((tensor_cols >> 16) & 0xFFFFu)        // tensor_dim0[31:16]
          | ((tensor_rows & 0xFFFFu) << 16));            // tensor_dim1[15:0]
    g1[3] = (int)(((tensor_rows >> 16) & 0xFFFFu)        // tensor_dim1[31:16]
          | ((tile_cols & 0xFFFFu) << 16));              // tile_dim0
    g1[4] = (int)(tile_rows & 0xFFFFu);                  // tile_dim1 | tile_dim2=0
    g1[5] = (int)row_stride;                             // tensor_dim0_stride[31:0]
    g1[6] = 0;                                           // stride hi | dim1_stride lo
    g1[7] = 0;                                           // dim1_stride hi

    v4i z4 = (v4i)0;
#if __clang_major__ >= 23
    v8i z8 = (v8i)0;
    __builtin_amdgcn_tensor_load_to_lds(g0, g1, z4, z4, z8, 0);
#else
    __builtin_amdgcn_tensor_load_to_lds(g0, g1, z4, z4, 0);
#endif
}

// ---------------------------------------------------------------------------
// Fragment loaders, wave32, contiguous k order (k = 16*(lane/16) + s) in A and B.
// A (16x32): lanes 0-15 row=lane, K=0..15; lanes 16-31 K=16..31
// B (32x16): lane&15=N column; lanes<16 K=0..15, lanes>=16 K=16..31 per lane
// C/D (16x16 f32): VGPR r -> (M=r,N=lane) / (M=r+8,N=lane-16)
// ---------------------------------------------------------------------------
static __device__ __forceinline__ v16h load_A_f16(const _Float16* A,
                                                  int lda, int kbase, int lane) {
    const int m  = lane & 15;
    const int kb = (lane & 16) ? 16 : 0;
    const _Float16* p = A + (size_t)m * lda + kbase + kb;
    v8h lo = *(const v8h*)(p);
    v8h hi = *(const v8h*)(p + 8);
    v16h a;
#pragma unroll
    for (int i = 0; i < 8; ++i) { a[i] = lo[i]; a[8 + i] = hi[i]; }
    return a;
}

static __device__ __forceinline__ v16h load_B_f16(const _Float16* M,
                                                  int ld, int col0, int kbase, int lane) {
    const int n  = lane & 15;
    const int kb = (lane & 16) ? 16 : 0;
    const _Float16* p = M + (size_t)(col0 + n) * ld + kbase + kb;
    return *(const v16h*)p;
}

// ---------------------------------------------------------------------------
// Kernel 0: f32 -> f16 conversion (4 elements/thread).
// ---------------------------------------------------------------------------
__global__ __launch_bounds__(256) void cvt_kernel(const float* __restrict__ src,
                                                  _Float16* __restrict__ dst, int n) {
    const int idx = (blockIdx.x * blockDim.x + threadIdx.x) * 4;
    if (idx + 3 < n) {
        v4f f = *(const v4f*)(src + idx);
        v8h h4 = {};
        h4[0] = (_Float16)f[0]; h4[1] = (_Float16)f[1];
        h4[2] = (_Float16)f[2]; h4[3] = (_Float16)f[3];
        dst[idx + 0] = h4[0]; dst[idx + 1] = h4[1];
        dst[idx + 2] = h4[2]; dst[idx + 3] = h4[3];
    }
}

// ---------------------------------------------------------------------------
// Kernel 1: QKV projection, TDM-fed. Block = 8 waves (4x2), tile 128x128.
// A/B K=64 slabs (128x64 f16 = 16KB each) double-buffered in LDS via TDM:
// 16 WMMAs per wave per barrier interval, 12 sync rounds total.
// Writes Q (scaled) [B,H,N,D], K [B,H,N,D], V^T [B,H,D,N], all f16.
// ---------------------------------------------------------------------------
__global__ __launch_bounds__(256) void qkv_kernel(const _Float16* __restrict__ xh,
                                                  const _Float16* __restrict__ wh,
                                                  _Float16* __restrict__ q,
                                                  _Float16* __restrict__ kbuf,
                                                  _Float16* __restrict__ vT) {
    __shared__ __align__(16) _Float16 As[2][128 * 64];   // 2 x 16KB
    __shared__ __align__(16) _Float16 Bs[2][128 * 64];   // 2 x 16KB

    const int lane = threadIdx.x & 31;
    const int wl   = threadIdx.x >> 5;
    const int NB   = E3 / 128;                 // 18 column blocks
    const int bm   = blockIdx.x / NB;          // 0..127
    const int bn   = blockIdx.x % NB;
    const int row0 = bm * 128;
    const int col0 = bn * 128;
    const int miw  = wl >> 1;                  // 0..3 -> 32-row slice
    const int niw  = wl & 1;                   // 0..1 -> 64-col slice

    if (wl == 0) {
        tdm_load_2d(&As[0][0], xh + (size_t)row0 * DIM, 64, 128, DIM, MTOK, DIM);
        tdm_load_2d(&Bs[0][0], wh + (size_t)col0 * DIM, 64, 128, DIM, E3, DIM);
    }

    v8f acc[2][4];
#pragma unroll
    for (int i = 0; i < 2; ++i)
#pragma unroll
        for (int j = 0; j < 4; ++j) acc[i][j] = (v8f)0.0f;

    for (int it = 0; it < DIM / 64; ++it) {
        if (wl == 0) __builtin_amdgcn_s_wait_tensorcnt(0);
        __syncthreads();   // buffer it&1 ready; all waves done reading (it+1)&1
        if (wl == 0 && it + 1 < DIM / 64) {
            const int kk = (it + 1) * 64;
            tdm_load_2d(&As[(it + 1) & 1][0], xh + (size_t)row0 * DIM + kk,
                        64, 128, DIM, MTOK, DIM);
            tdm_load_2d(&Bs[(it + 1) & 1][0], wh + (size_t)col0 * DIM + kk,
                        64, 128, DIM, E3, DIM);
        }
        const _Float16* at = &As[it & 1][0];
        const _Float16* bt = &Bs[it & 1][0];
#pragma unroll
        for (int kc = 0; kc < 2; ++kc) {
            v16h a0 = load_A_f16(at + (miw * 32) * 64,      64, kc * 32, lane);
            v16h a1 = load_A_f16(at + (miw * 32 + 16) * 64, 64, kc * 32, lane);
#pragma unroll
            for (int j = 0; j < 4; ++j) {
                v16h b = load_B_f16(bt, 64, niw * 64 + j * 16, kc * 32, lane);
                acc[0][j] = wmma16(a0, b, acc[0][j]);
                acc[1][j] = wmma16(a1, b, acc[1][j]);
            }
        }
    }

    const int tn    = (col0 + niw * 64) / 64;  // 64-wide tile = one (which,head)
    const int which = tn / NH;
    const int h     = tn % NH;
    const int wrow0 = row0 + miw * 32;
    const int nloc  = lane & 15;
    const int rofs  = (lane & 16) ? 8 : 0;
#pragma unroll
    for (int mi = 0; mi < 2; ++mi) {
#pragma unroll
        for (int j = 0; j < 4; ++j) {
#pragma unroll
            for (int r = 0; r < 8; ++r) {
                const int m    = wrow0 + mi * 16 + rofs + r;
                const int d    = j * 16 + nloc;
                const int b    = m >> 10;
                const int nseq = m & 1023;
                const float v  = acc[mi][j][r];
                if (which == 0) {
                    q[(((size_t)(b * NH + h)) * SEQ + nseq) * HD + d] =
                        (_Float16)(v * QK_SCALE);
                } else if (which == 1) {
                    kbuf[(((size_t)(b * NH + h)) * SEQ + nseq) * HD + d] = (_Float16)v;
                } else {
                    vT[(((size_t)(b * NH + h)) * HD + d) * SEQ + nseq] = (_Float16)v;
                }
            }
        }
    }
}

// ---------------------------------------------------------------------------
// Kernel 2: fused flash attention, TDM-fed K/V tiles.
// Block = 8 waves sharing one (b,h); wave w owns q-rows [qt*16, qt*16+16).
// K tile [64 keys][64 d] and V^T tile [64 d][64 keys] double-buffered in LDS.
// S stays in registers; online softmax via half-wave shfl_xor; P re-swizzled
// C/D->A layout through per-wave LDS scratch.
// ---------------------------------------------------------------------------
__global__ __launch_bounds__(256) void attn_kernel(const _Float16* __restrict__ q,
                                                   const _Float16* __restrict__ kbuf,
                                                   const _Float16* __restrict__ vT,
                                                   _Float16* __restrict__ o) {
    __shared__ __align__(16) _Float16 Ks[2][64 * 64];      // 2 x 8KB
    __shared__ __align__(16) _Float16 Vs[2][64 * 64];      // 2 x 8KB
    __shared__ __align__(32) _Float16 pshare[8][16 * 64];  // 2KB per wave

    const int lane = threadIdx.x & 31;
    const int wl   = threadIdx.x >> 5;
    const int bh   = blockIdx.x >> 3;                 // 0..191  (b*12+h)
    const int qt   = (blockIdx.x & 7) * 8 + wl;       // 0..63
    const int row0 = qt * 16;

    const _Float16* Qp = q    + (size_t)bh * SEQ * HD;
    const _Float16* Kp = kbuf + (size_t)bh * SEQ * HD;
    const _Float16* Vp = vT   + (size_t)bh * HD * SEQ;
    _Float16* pb = &pshare[wl][0];

    if (wl == 0) {
        tdm_load_2d(&Ks[0][0], Kp, 64, 64, HD, SEQ, HD);    // keys 0..63
        tdm_load_2d(&Vs[0][0], Vp, 64, 64, SEQ, HD, SEQ);   // d rows, keys 0..63
    }

    v16h qa0 = load_A_f16(Qp + (size_t)row0 * HD, HD, 0,  lane);
    v16h qa1 = load_A_f16(Qp + (size_t)row0 * HD, HD, 32, lane);

    v8f oacc[4];
#pragma unroll
    for (int t = 0; t < 4; ++t) oacc[t] = (v8f)0.0f;
    float mr[8], lr[8];
#pragma unroll
    for (int r = 0; r < 8; ++r) { mr[r] = -3.0e38f; lr[r] = 0.0f; }

    const int nloc = lane & 15;
    const int rofs = (lane & 16) ? 8 : 0;

    for (int it = 0; it < SEQ / 64; ++it) {
        if (wl == 0) __builtin_amdgcn_s_wait_tensorcnt(0);
        __syncthreads();
        if (wl == 0 && it + 1 < SEQ / 64) {
            const int kb1 = (it + 1) * 64;
            tdm_load_2d(&Ks[(it + 1) & 1][0], Kp + (size_t)kb1 * HD, 64, 64, HD, SEQ, HD);
            tdm_load_2d(&Vs[(it + 1) & 1][0], Vp + kb1, 64, 64, SEQ, HD, SEQ);
        }
        const _Float16* kt = &Ks[it & 1][0];
        const _Float16* vt = &Vs[it & 1][0];

        // ---- S = Q * K^T for 64 keys (4 tiles of 16) ----
        v8f s[4];
#pragma unroll
        for (int j = 0; j < 4; ++j) {
            v16h b0 = load_B_f16(kt, HD, j * 16, 0,  lane);
            v16h b1 = load_B_f16(kt, HD, j * 16, 32, lane);
            s[j] = wmma16(qa0, b0, (v8f)0.0f);
            s[j] = wmma16(qa1, b1, s[j]);
        }

        // ---- online softmax ----
#pragma unroll
        for (int r = 0; r < 8; ++r) {
            float vmax = fmaxf(fmaxf(s[0][r], s[1][r]), fmaxf(s[2][r], s[3][r]));
#pragma unroll
            for (int off = 8; off >= 1; off >>= 1)
                vmax = fmaxf(vmax, __shfl_xor(vmax, off, 32));
            const float mnew  = fmaxf(mr[r], vmax);
            const float alpha = __expf(mr[r] - mnew);
            mr[r] = mnew;
            float rsum = 0.0f;
#pragma unroll
            for (int j = 0; j < 4; ++j) {
                const float pv = __expf(s[j][r] - mnew);
                s[j][r] = pv;
                rsum += pv;
            }
#pragma unroll
            for (int off = 8; off >= 1; off >>= 1)
                rsum += __shfl_xor(rsum, off, 32);
            lr[r] = lr[r] * alpha + rsum;
#pragma unroll
            for (int t = 0; t < 4; ++t) oacc[t][r] *= alpha;
        }

        // ---- spill P (C/D layout) -> LDS row-major [16][64] ----
#pragma unroll
        for (int j = 0; j < 4; ++j)
#pragma unroll
            for (int r = 0; r < 8; ++r)
                pb[(rofs + r) * 64 + j * 16 + nloc] = (_Float16)s[j][r];
        asm volatile("s_wait_dscnt 0x0" ::: "memory");  // intra-wave DS ordering

        // ---- O += P * V ----
#pragma unroll
        for (int kc = 0; kc < 2; ++kc) {
            v16h pa = load_A_f16(pb, 64, kc * 32, lane);
#pragma unroll
            for (int t = 0; t < 4; ++t) {
                v16h vb = load_B_f16(vt, 64, t * 16, kc * 32, lane);
                oacc[t] = wmma16(pa, vb, oacc[t]);
            }
        }
    }

    float invl[8];
#pragma unroll
    for (int r = 0; r < 8; ++r) invl[r] = 1.0f / lr[r];
    const int bb = bh / NH;
    const int h  = bh % NH;
#pragma unroll
    for (int t = 0; t < 4; ++t)
#pragma unroll
        for (int r = 0; r < 8; ++r) {
            const int tok = row0 + rofs + r;
            o[((size_t)(bb * SEQ + tok)) * DIM + h * HD + t * 16 + nloc] =
                (_Float16)(oacc[t][r] * invl[r]);
        }
}

// ---------------------------------------------------------------------------
// Kernel 3: output projection, TDM-fed, same K=64 slab structure as qkv.
// out[m,o] = sum_c O[m,c] * w_proj[o,c] + b_proj[o], f32 out.
// ---------------------------------------------------------------------------
__global__ __launch_bounds__(256) void proj_kernel(const _Float16* __restrict__ oh,
                                                   const _Float16* __restrict__ ph,
                                                   const float* __restrict__ bproj,
                                                   float* __restrict__ out) {
    __shared__ __align__(16) _Float16 As[2][128 * 64];
    __shared__ __align__(16) _Float16 Bs[2][128 * 64];

    const int lane = threadIdx.x & 31;
    const int wl   = threadIdx.x >> 5;
    const int NB   = DIM / 128;                // 6
    const int bm   = blockIdx.x / NB;
    const int bn   = blockIdx.x % NB;
    const int row0 = bm * 128;
    const int col0 = bn * 128;
    const int miw  = wl >> 1;
    const int niw  = wl & 1;

    if (wl == 0) {
        tdm_load_2d(&As[0][0], oh + (size_t)row0 * DIM, 64, 128, DIM, MTOK, DIM);
        tdm_load_2d(&Bs[0][0], ph + (size_t)col0 * DIM, 64, 128, DIM, DIM, DIM);
    }

    v8f acc[2][4];
#pragma unroll
    for (int i = 0; i < 2; ++i)
#pragma unroll
        for (int j = 0; j < 4; ++j) acc[i][j] = (v8f)0.0f;

    for (int it = 0; it < DIM / 64; ++it) {
        if (wl == 0) __builtin_amdgcn_s_wait_tensorcnt(0);
        __syncthreads();
        if (wl == 0 && it + 1 < DIM / 64) {
            const int kk = (it + 1) * 64;
            tdm_load_2d(&As[(it + 1) & 1][0], oh + (size_t)row0 * DIM + kk,
                        64, 128, DIM, MTOK, DIM);
            tdm_load_2d(&Bs[(it + 1) & 1][0], ph + (size_t)col0 * DIM + kk,
                        64, 128, DIM, DIM, DIM);
        }
        const _Float16* at = &As[it & 1][0];
        const _Float16* bt = &Bs[it & 1][0];
#pragma unroll
        for (int kc = 0; kc < 2; ++kc) {
            v16h a0 = load_A_f16(at + (miw * 32) * 64,      64, kc * 32, lane);
            v16h a1 = load_A_f16(at + (miw * 32 + 16) * 64, 64, kc * 32, lane);
#pragma unroll
            for (int j = 0; j < 4; ++j) {
                v16h b = load_B_f16(bt, 64, niw * 64 + j * 16, kc * 32, lane);
                acc[0][j] = wmma16(a0, b, acc[0][j]);
                acc[1][j] = wmma16(a1, b, acc[1][j]);
            }
        }
    }

    const int nloc = lane & 15;
    const int rofs = (lane & 16) ? 8 : 0;
    const int wrow0 = row0 + miw * 32;
    const int wcol0 = col0 + niw * 64;
#pragma unroll
    for (int mi = 0; mi < 2; ++mi)
#pragma unroll
        for (int j = 0; j < 4; ++j) {
            const int   col  = wcol0 + j * 16 + nloc;
            const float bias = bproj[col];
#pragma unroll
            for (int r = 0; r < 8; ++r) {
                const int m = wrow0 + mi * 16 + rofs + r;
                out[(size_t)m * DIM + col] = acc[mi][j][r] + bias;
            }
        }
}

// ---------------------------------------------------------------------------
// Launcher. Workspace layout (bytes):
//   [0, SEG)        xh   (x in f16)
//   [SEG, 2SEG)     Q    [B,H,N,D] f16 (pre-scaled)
//   [2SEG, 3SEG)    K    [B,H,N,D] f16
//   [3SEG, 4SEG)    V^T  [B,H,D,N] f16
//   [4SEG, 5SEG)    O    [B,N,C]   f16
//   then w_qkv f16 (3538944 B), w_proj f16 (1179648 B).  Total ~130.5 MB.
// ---------------------------------------------------------------------------
extern "C" void kernel_launch(void* const* d_in, const int* in_sizes, int n_in,
                              void* d_out, int out_size, void* d_ws, size_t ws_size,
                              hipStream_t stream) {
    (void)in_sizes; (void)n_in; (void)out_size; (void)ws_size;
    const float* x     = (const float*)d_in[0];
    const float* wqkv  = (const float*)d_in[1];
    const float* wproj = (const float*)d_in[2];
    const float* bproj = (const float*)d_in[3];
    float* out = (float*)d_out;

    const size_t SEG = (size_t)MTOK * DIM * sizeof(_Float16);  // 25165824
    char* ws = (char*)d_ws;
    _Float16* xh = (_Float16*)(ws);
    _Float16* q  = (_Float16*)(ws + SEG);
    _Float16* k  = (_Float16*)(ws + 2 * SEG);
    _Float16* vT = (_Float16*)(ws + 3 * SEG);
    _Float16* oh = (_Float16*)(ws + 4 * SEG);
    _Float16* wh = (_Float16*)(ws + 5 * SEG);
    _Float16* ph = (_Float16*)(ws + 5 * SEG + (size_t)E3 * DIM * sizeof(_Float16));

    // f32 -> f16 conversions (4 elems/thread)
    cvt_kernel<<<(MTOK * DIM) / 1024, 256, 0, stream>>>(x, xh, MTOK * DIM);
    cvt_kernel<<<(E3 * DIM) / 1024,   256, 0, stream>>>(wqkv, wh, E3 * DIM);
    cvt_kernel<<<(DIM * DIM) / 1024,  256, 0, stream>>>(wproj, ph, DIM * DIM);

    // qkv: (16384/128)*(2304/128) = 128*18 = 2304 blocks of 8 waves
    qkv_kernel<<<2304, 256, 0, stream>>>(xh, wh, q, k, vT);
    // attn: 192 (b,h) * 8 q-tile groups = 1536 blocks
    attn_kernel<<<1536, 256, 0, stream>>>(q, k, vT, oh);
    // proj: (16384/128)*(768/128) = 128*6 = 768 blocks
    proj_kernel<<<768, 256, 0, stream>>>(oh, ph, bproj, out);
}
